// FilterDetection_52235392254189
// MI455X (gfx1250) — matrix operand order repaired and checked by reading, not compiled, not run on hardware
//
#include <hip/hip_runtime.h>
#include <hip/hip_bf16.h>

typedef __attribute__((ext_vector_type(16))) _Float16 v16h;
typedef __attribute__((ext_vector_type(8)))  float    v8f;

// Problem constants (match reference)
#define BB 2
#define AA 100000
#define CC 80
#define NC 79          /* classes excluding background (IGNORE=0) */
#define PERF 500
#define PAD 512        /* PERF padded to power of two */
#define PROP 100
#define IOU_THR 0.5f
#define SCORE_THR 0.05f
#define MAX_RATIO 4.135166556742356f  /* |log(16/1000)| */

__device__ __forceinline__ float clip01(float x) {
    return fminf(fmaxf(x, 0.0f), 1.0f);
}

// ---------------------------------------------------------------------------
// Kernel 1: decode deltas against anchors, clip to [0,1].  boxes: [B,A,4]
// ---------------------------------------------------------------------------
__global__ void fd_decode_kernel(const float* __restrict__ bbox_pred,
                                 const float* __restrict__ anchors,
                                 float* __restrict__ boxes) {
    int i = blockIdx.x * blockDim.x + threadIdx.x;   // over B*A
    if (i >= BB * AA) return;
    int a = i % AA;
    float4 d  = ((const float4*)bbox_pred)[i];
    float4 an = ((const float4*)anchors)[a];
    float dx = d.x, dy = d.y;
    float dw = fminf(fmaxf(d.z, -MAX_RATIO), MAX_RATIO);
    float dh = fminf(fmaxf(d.w, -MAX_RATIO), MAX_RATIO);
    float pw = an.z - an.x, ph = an.w - an.y;
    float px = (an.x + an.z) * 0.5f, py = (an.y + an.w) * 0.5f;
    float gw = pw * __expf(dw), gh = ph * __expf(dh);
    float gx = px + pw * dx,    gy = py + ph * dy;
    float4 o;
    o.x = clip01(gx - gw * 0.5f);
    o.y = clip01(gy - gh * 0.5f);
    o.z = clip01(gx + gw * 0.5f);
    o.w = clip01(gy + gh * 0.5f);
    ((float4*)boxes)[i] = o;
}

// ---------------------------------------------------------------------------
// Kernel 2: per (batch, class) top-500 selection (padded to 512) with bitonic sort.
// One block per (b,ci).  Histogram cutoff -> compact -> sort -> emit.
// ---------------------------------------------------------------------------
__global__ void fd_class_topk_kernel(const float* __restrict__ y_pred,
                                     float* __restrict__ tk_val,
                                     int*   __restrict__ tk_idx) {
    int blk = blockIdx.x;            // 0 .. B*NC-1
    int b = blk / NC, ci = blk % NC;
    int cls = ci + 1;                // skip background column 0
    const float* sc = y_pred + (size_t)b * AA * CC + cls;

    __shared__ unsigned hist[1024];
    __shared__ float    cv[1024];
    __shared__ int      cidx[1024];
    __shared__ unsigned cnt;
    __shared__ int      cutoff;

    for (int i = threadIdx.x; i < 1024; i += blockDim.x) hist[i] = 0;
    if (threadIdx.x == 0) cnt = 0;
    __syncthreads();

    // Pass 1: value histogram (bin monotone in score).  Prefetch ahead.
    for (int a = threadIdx.x; a < AA; a += blockDim.x) {
        if (a + 4096 < AA)
            __builtin_prefetch(sc + (size_t)(a + 4096) * CC, 0, 1);
        float s = sc[(size_t)a * CC];
        if (s > SCORE_THR) {
            int bin = (int)fminf(s * 1024.0f, 1023.0f);
            atomicAdd(&hist[bin], 1u);
        }
    }
    __syncthreads();

    if (threadIdx.x == 0) {
        unsigned acc = 0; int cb = 0;
        for (int bin = 1023; bin >= 0; --bin) {
            acc += hist[bin];
            if (acc >= PERF) { cb = bin; break; }
        }
        cutoff = cb;
    }
    __syncthreads();
    int cb = cutoff;

    // Pass 2: compact candidates at/above cutoff bin.
    for (int a = threadIdx.x; a < AA; a += blockDim.x) {
        float s = sc[(size_t)a * CC];
        if (s > SCORE_THR) {
            int bin = (int)fminf(s * 1024.0f, 1023.0f);
            if (bin >= cb) {
                unsigned slot = atomicAdd(&cnt, 1u);
                if (slot < 1024u) { cv[slot] = s; cidx[slot] = a; }
            }
        }
    }
    __syncthreads();
    unsigned n = cnt < 1024u ? cnt : 1024u;
    for (int i = threadIdx.x; i < 1024; i += blockDim.x)
        if ((unsigned)i >= n) { cv[i] = -1e30f; cidx[i] = 0x7fffffff; }
    __syncthreads();

    // Bitonic sort, descending by (val desc, idx asc).
    for (int k = 2; k <= 1024; k <<= 1) {
        for (int j = k >> 1; j > 0; j >>= 1) {
            for (int i = threadIdx.x; i < 1024; i += blockDim.x) {
                int p = i ^ j;
                if (p > i) {
                    bool desc = ((i & k) == 0);
                    float vi = cv[i], vp = cv[p];
                    int   ii = cidx[i], ip = cidx[p];
                    bool greater = (vi > vp) || (vi == vp && ii < ip);
                    if (desc != greater) {
                        cv[i] = vp; cv[p] = vi;
                        cidx[i] = ip; cidx[p] = ii;
                    }
                }
            }
            __syncthreads();
        }
    }

    float* ov = tk_val + (size_t)blk * PAD;
    int*   oi = tk_idx + (size_t)blk * PAD;
    for (int r = threadIdx.x; r < PAD; r += blockDim.x) {
        float v = (r < PERF && cv[r] > 0.0f) ? cv[r] : -1.0f;
        ov[r] = v;
        oi[r] = (cidx[r] == 0x7fffffff) ? 0 : cidx[r];
    }
}

// ---------------------------------------------------------------------------
// Kernel 3: greedy NMS per (batch,class).  Adjacency bitmask in LDS; the
// cross-tile suppression matvec runs on v_wmma_f32_16x16x32_f16 (wave 0).
// ---------------------------------------------------------------------------
__global__ void fd_nms_kernel(const float* __restrict__ boxes,
                              float* __restrict__ tk_val,
                              const int* __restrict__ tk_idx) {
    int blk = blockIdx.x;            // 0 .. B*NC-1
    int b = blk / NC;

    __shared__ float4   bx[PAD];           // 8 KB
    __shared__ unsigned adj[PAD * 16];     // 32 KB: row i, 16 words of col bits
    __shared__ _Float16 kept_h[PAD];       // 1 KB
    __shared__ float    vals[PAD];         // 2 KB

    float*      tv = tk_val + (size_t)blk * PAD;
    const int*  ti = tk_idx + (size_t)blk * PAD;

    for (int r = threadIdx.x; r < PAD; r += blockDim.x) {
        vals[r] = tv[r];
        int aidx = ti[r];
        bx[r] = ((const float4*)boxes)[(size_t)b * AA + aidx];
        kept_h[r] = (_Float16)0.0f;
    }
    __syncthreads();

    // Adjacency bitmask: adj[i*16+w] bit t  <=>  IoU(box_i, box_{32w+t}) > thr
    for (int w = threadIdx.x; w < PAD * 16; w += blockDim.x) {
        int i  = w >> 4;
        int wc = w & 15;
        float4 bi = bx[i];
        float ai = fmaxf(bi.z - bi.x, 0.0f) * fmaxf(bi.w - bi.y, 0.0f);
        unsigned bits = 0;
        #pragma unroll 8
        for (int t2 = 0; t2 < 32; ++t2) {
            int j = (wc << 5) + t2;
            float4 bj = bx[j];
            float xx1 = fmaxf(bi.x, bj.x), yy1 = fmaxf(bi.y, bj.y);
            float xx2 = fminf(bi.z, bj.z), yy2 = fminf(bi.w, bj.w);
            float iw = fmaxf(xx2 - xx1, 0.0f), ih = fmaxf(yy2 - yy1, 0.0f);
            float inter = iw * ih;
            float aj = fmaxf(bj.z - bj.x, 0.0f) * fmaxf(bj.w - bj.y, 0.0f);
            float uni = fmaxf(ai + aj - inter, 1e-9f);
            if (inter > IOU_THR * uni) bits |= 1u << t2;
        }
        adj[w] = bits;
    }
    __syncthreads();

    // Greedy scan: wave 0 only (EXEC all-1s across the 32 lanes, as WMMA needs).
    if (threadIdx.x < 32) {
        int lane = threadIdx.x;
        int nn   = lane & 15;                    // output column within tile
        for (int t = 0; t < PAD / 16; ++t) {
            int i0 = t * 16;
            // Cross-tile suppression counts s[i0+n] = sum_{j<i0} kept[j]*adj[j,i0+n]
            v8f acc = {};
            int col  = i0 + nn;
            int wsel = col >> 5, bsel = col & 31;
            for (int kc = 0; kc < i0; kc += 32) {
                v16h Ah, Bh;
                // A (16x32 f16, k = e + 16*(lane/16)): row 0 = kept flags, rows 1-15 = 0
                #pragma unroll
                for (int e = 0; e < 16; ++e) {
                    int K = e + ((lane < 16) ? 0 : 16);
                    float kv = ((lane & 15) == 0) ? (float)kept_h[kc + K] : 0.0f;
                    Ah[e] = (_Float16)kv;
                }
                // B (32x16 f16): element (K=jb+e, N=nn) from adjacency bitmask
                int jb = kc + ((lane < 16) ? 0 : 16);
                #pragma unroll
                for (int e = 0; e < 16; ++e) {
                    unsigned bits = adj[(jb + e) * 16 + wsel];
                    Bh[e] = (_Float16)(float)((bits >> bsel) & 1u);
                }
                acc = __builtin_amdgcn_wmma_f32_16x16x32_f16(
                    false, Ah, false, Bh, (short)0, acc, false, false);
            }
            float s_n = acc[0];   // D row M=0: lanes 0-15 hold columns i0..i0+15

            // Intra-tile 16x16 adjacency column mask for column nn
            unsigned lm = 0;
            #pragma unroll
            for (int m = 0; m < 16; ++m) {
                unsigned bits = adj[(i0 + m) * 16 + wsel];
                lm |= ((bits >> bsel) & 1u) << m;
            }
            int vf = (vals[i0 + nn] > 0.0f) ? 1 : 0;

            // 16-step serial greedy within the tile (redundant on all lanes)
            unsigned keptmask = 0;
            for (int m = 0; m < 16; ++m) {
                float    sm = __shfl(s_n, m, 32);
                unsigned am = __shfl(lm, m, 32);
                int      fm = __shfl(vf, m, 32);
                bool sup = (sm > 0.5f) || ((am & keptmask) != 0u);
                if (fm && !sup) keptmask |= (1u << m);
            }
            if (lane < 16) {
                bool k = (keptmask >> nn) & 1u;
                kept_h[i0 + nn] = (_Float16)(k ? 1.0f : 0.0f);
                tv[i0 + nn] = k ? vals[i0 + nn] : -1.0f;
            }
        }
    }
}

// ---------------------------------------------------------------------------
// Kernel 4: global top-PROP per image via iterative argmax over [NC*PAD].
// tk_val is regenerated every launch, so mutating it is replay-safe.
// ---------------------------------------------------------------------------
__global__ void fd_global_topk_kernel(float* __restrict__ tk_val,
                                      float* __restrict__ sel_val,
                                      int*   __restrict__ sel_pos) {
    int b = blockIdx.x;
    float* vals = tk_val + (size_t)b * NC * PAD;
    const int N = NC * PAD;
    __shared__ float smax[256];
    __shared__ int   spos[256];
    for (int it = 0; it < PROP; ++it) {
        float mv = -1e30f; int mp = N;
        for (int i = threadIdx.x; i < N; i += blockDim.x) {
            float v = vals[i];
            if (v > mv || (v == mv && i < mp)) { mv = v; mp = i; }
        }
        smax[threadIdx.x] = mv; spos[threadIdx.x] = mp;
        __syncthreads();
        for (int off = 128; off > 0; off >>= 1) {
            if ((int)threadIdx.x < off) {
                float v2 = smax[threadIdx.x + off];
                int   p2 = spos[threadIdx.x + off];
                if (v2 > smax[threadIdx.x] ||
                    (v2 == smax[threadIdx.x] && p2 < spos[threadIdx.x])) {
                    smax[threadIdx.x] = v2; spos[threadIdx.x] = p2;
                }
            }
            __syncthreads();
        }
        if (threadIdx.x == 0) {
            sel_val[b * PROP + it] = smax[0];
            sel_pos[b * PROP + it] = spos[0];
            if (spos[0] < N) vals[spos[0]] = -2e30f;
        }
        __syncthreads();
    }
}

// ---------------------------------------------------------------------------
// Kernel 5: gather final scores/boxes.  out = [B,PROP,C] scores ++ [B,PROP,4]
// ---------------------------------------------------------------------------
__global__ void fd_gather_kernel(const float* __restrict__ y_pred,
                                 const float* __restrict__ boxes,
                                 const float* __restrict__ sel_val,
                                 const int*   __restrict__ sel_pos,
                                 const int*   __restrict__ tk_idx,
                                 float* __restrict__ out) {
    int bp = blockIdx.x;             // 0 .. B*PROP-1
    int b = bp / PROP;
    float v = sel_val[bp];
    bool valid = v > 0.0f;
    int pos = sel_pos[bp];
    if (pos >= NC * PAD) { pos = 0; valid = false; }
    int ci = pos / PAD, r = pos % PAD;
    int aidx = tk_idx[((size_t)b * NC + ci) * PAD + r];

    float* os = out + (size_t)bp * CC;
    float* ob = out + (size_t)BB * PROP * CC + (size_t)bp * 4;
    const float* srow = y_pred + ((size_t)b * AA + aidx) * CC;
    float4 bxv = ((const float4*)boxes)[(size_t)b * AA + aidx];

    for (int c = threadIdx.x; c < CC; c += blockDim.x)
        os[c] = valid ? srow[c] : 0.0f;
    if (threadIdx.x == 0) {
        ob[0] = valid ? bxv.x : 0.0f;
        ob[1] = valid ? bxv.y : 0.0f;
        ob[2] = valid ? bxv.z : 0.0f;
        ob[3] = valid ? bxv.w : 0.0f;
    }
}

// ---------------------------------------------------------------------------
extern "C" void kernel_launch(void* const* d_in, const int* in_sizes, int n_in,
                              void* d_out, int out_size, void* d_ws, size_t ws_size,
                              hipStream_t stream) {
    const float* y_pred    = (const float*)d_in[0];   // [B,A,C]
    const float* bbox_pred = (const float*)d_in[1];   // [B,A,4]
    const float* anchors   = (const float*)d_in[2];   // [A,4]
    float* out = (float*)d_out;

    // Workspace layout (bytes)
    char* ws = (char*)d_ws;
    size_t off = 0;
    float* boxes  = (float*)(ws + off); off += (size_t)BB * AA * 4 * sizeof(float);     // 3.2 MB
    float* tk_val = (float*)(ws + off); off += (size_t)BB * NC * PAD * sizeof(float);   // 316 KB
    int*   tk_idx = (int*)  (ws + off); off += (size_t)BB * NC * PAD * sizeof(int);     // 316 KB
    float* sel_val= (float*)(ws + off); off += (size_t)BB * PROP * sizeof(float);
    int*   sel_pos= (int*)  (ws + off); off += (size_t)BB * PROP * sizeof(int);
    (void)ws_size; (void)in_sizes; (void)n_in; (void)out_size;

    fd_decode_kernel<<<(BB * AA + 255) / 256, 256, 0, stream>>>(bbox_pred, anchors, boxes);
    fd_class_topk_kernel<<<BB * NC, 256, 0, stream>>>(y_pred, tk_val, tk_idx);
    fd_nms_kernel<<<BB * NC, 256, 0, stream>>>(boxes, tk_val, tk_idx);
    fd_global_topk_kernel<<<BB, 256, 0, stream>>>(tk_val, sel_val, sel_pos);
    fd_gather_kernel<<<BB * PROP, 128, 0, stream>>>(y_pred, boxes, sel_val, sel_pos, tk_idx, out);
}
